// GraphNet_41944650613153
// MI455X (gfx1250) — matrix operand order repaired and checked by reading, not compiled, not run on hardware
//
#include <hip/hip_runtime.h>

// ---------------------------------------------------------------------------
// GCN 3-layer fused kernel for gfx1250 (MI455X), wave32, WMMA bf16.
//   layer: H = A_in @ W ; AG = Adj @ H ; A_next = relu(bn(AG + b))
// Sparse aggregation is converted to a dense [160x160] adjacency GEMM.
// Everything for one batch element lives in LDS (~272KB of the 320KB/WGP).
// ---------------------------------------------------------------------------

typedef __attribute__((ext_vector_type(16))) __bf16 v16bf;
typedef __attribute__((ext_vector_type(8)))  float  v8f;

#define NN     156        // nodes
#define EE     1564       // edges (before self loops)
#define CIN    301        // input channels
#define CH     128        // hidden/out channels
#define MROWS  160        // padded node rows (multiple of 16/32)
#define K0P    320        // padded K for layer0 (301 -> 320, multiple of 32)
#define KAGG   160        // K of aggregation GEMM (= MROWS)
#define HT_LD  168        // element stride of transposed-H rows (16B aligned, de-conflicted)

// LDS byte offsets (all 16B aligned)
#define OFF_M  0                       // adjacency bf16 [160][160]        = 51200 B (persistent)
#define OFF_A  51200                   // activations bf16, max [160][320] = 102400 B
#define OFF_W  153600                  // W^T bf16, max [128][320]         = 81920 B
#define OFF_H  235520                  // H^T bf16 [128][168]              = 43008 B
#define LDS_BYTES 278528               // < 320KB

union Frag { v16bf bf; float4 f4[2]; };

__device__ __forceinline__ unsigned short f2bf(float x) {
  union { float f; unsigned u; } v; v.f = x;
  unsigned r = v.u + 0x7FFFu + ((v.u >> 16) & 1u);   // round-to-nearest-even
  return (unsigned short)(r >> 16);
}

__device__ __forceinline__ unsigned pack2bf(float lo, float hi) {
  return (unsigned)f2bf(lo) | ((unsigned)f2bf(hi) << 16);
}

// ---------------------------------------------------------------------------
// Generic 160x128 GEMM on WMMA, 2x2 register tiling per wave (8 waves, 20 quads).
//   A: bf16 [160][kpad]   row-major, stride ldA bytes  (WMMA A layout via 2x b128)
//   B: bf16 stored transposed: row n = output column, stride ldB bytes
// Epilogues (compile-time):
//   0: store D transposed bf16 into OFF_H ([c][s], stride HT_LD)  -- feeds agg GEMM
//   1: D = relu((D + cb)*bnw/sqrt(1+eps) + bnb) -> bf16 A_next at OFF_A ([d][128])
//   2: D = D + cb -> fp32 global (rows < NN only)
// ---------------------------------------------------------------------------
template <int EPI>
__device__ __forceinline__ void wmma_gemm(
    char* smem, int offA, int ldA, int offB, int ldB, int kpad,
    const float* __restrict__ bnw, const float* __restrict__ bnb,
    const float* __restrict__ cb, float* __restrict__ gout)
{
  const int tid  = threadIdx.x;
  const int wave = tid >> 5;
  const int lane = tid & 31;
  const int half = lane >> 4;
  const int l16  = lane & 15;

  for (int q = wave; q < 20; q += 8) {            // 5 x 4 super-tiles of 32x32
    const int m0 = (q >> 2) << 5;                 // 0..128
    const int n0 = (q & 3) << 5;                  // 0..96
    v8f acc00 = {0.f,0.f,0.f,0.f,0.f,0.f,0.f,0.f};
    v8f acc01 = acc00, acc10 = acc00, acc11 = acc00;

    for (int k0 = 0; k0 < kpad; k0 += 32) {
      Frag a0, a1, b0, b1;
      // A 16x32 bf16 fragment: lane half selects K blocks {0-7,16-23} / {8-15,24-31}
      {
        const char* p  = smem + offA + (m0 + l16) * ldA + ((k0 + 8 * half) << 1);
        a0.f4[0] = *(const float4*)p;        a0.f4[1] = *(const float4*)(p + 32);
        const char* p2 = p + 16 * ldA;
        a1.f4[0] = *(const float4*)p2;       a1.f4[1] = *(const float4*)(p2 + 32);
      }
      // B 32x16 bf16 fragment from transposed rows: lane half selects K blocks {0-7,16-23} / {8-15,24-31}
      {
        const char* p  = smem + offB + (n0 + l16) * ldB + ((k0 + 8 * half) << 1);
        b0.f4[0] = *(const float4*)p;        b0.f4[1] = *(const float4*)(p + 32);
        const char* p2 = p + 16 * ldB;
        b1.f4[0] = *(const float4*)p2;       b1.f4[1] = *(const float4*)(p2 + 32);
      }
      acc00 = __builtin_amdgcn_wmma_f32_16x16x32_bf16(false, a0.bf, false, b0.bf, (short)0, acc00, false, false);
      acc01 = __builtin_amdgcn_wmma_f32_16x16x32_bf16(false, a0.bf, false, b1.bf, (short)0, acc01, false, false);
      acc10 = __builtin_amdgcn_wmma_f32_16x16x32_bf16(false, a1.bf, false, b0.bf, (short)0, acc10, false, false);
      acc11 = __builtin_amdgcn_wmma_f32_16x16x32_bf16(false, a1.bf, false, b1.bf, (short)0, acc11, false, false);
    }

    v8f accs[2][2] = {{acc00, acc01}, {acc10, acc11}};
#pragma unroll
    for (int i = 0; i < 2; ++i) {
#pragma unroll
      for (int j = 0; j < 2; ++j) {
        const int mb = m0 + 16 * i;
        const int c  = n0 + 16 * j + l16;          // D column for this lane
        if constexpr (EPI == 0) {
          unsigned short* ht = (unsigned short*)(smem + OFF_H);
#pragma unroll
          for (int r = 0; r < 8; ++r) {
            const int s = mb + r + 8 * half;       // D row (M) per ISA C/D layout
            ht[c * HT_LD + s] = f2bf(accs[i][j][r]);
          }
        } else if constexpr (EPI == 1) {
          const float scale = bnw[c] * 0.9999950000374997f;   // 1/sqrt(1+1e-5)
          const float bb    = bnb[c];
          const float cbias = cb[c];
          unsigned short* an = (unsigned short*)(smem + OFF_A);
#pragma unroll
          for (int r = 0; r < 8; ++r) {
            const int d = mb + r + 8 * half;
            const float v = (accs[i][j][r] + cbias) * scale + bb;
            an[d * CH + c] = f2bf(fmaxf(v, 0.f));
          }
        } else {
          const float cbias = cb[c];
#pragma unroll
          for (int r = 0; r < 8; ++r) {
            const int d = mb + r + 8 * half;
            if (d < NN) gout[d * CH + c] = accs[i][j][r] + cbias;
          }
        }
      }
    }
  }
}

// ---------------------------------------------------------------------------
// Staging helpers: 4-wide along K, one index computation per 4 elements,
// packed bf16x2 dwords, single ds_store_b64 per chunk.
// ---------------------------------------------------------------------------
__device__ __forceinline__ void stage_wt128(char* smem, const float* __restrict__ W) {
  unsigned short* Wt = (unsigned short*)(smem + OFF_W);   // [128][128] bf16, W^T
  const int tid = threadIdx.x;
  for (int i = tid; i < CH * (CH / 4); i += 256) {
    const int n  = i >> 5;               // output column
    const int k0 = (i & 31) << 2;
    uint2 pk;
    pk.x = pack2bf(W[(k0 + 0) * CH + n], W[(k0 + 1) * CH + n]);
    pk.y = pack2bf(W[(k0 + 2) * CH + n], W[(k0 + 3) * CH + n]);
    *(uint2*)(Wt + n * CH + k0) = pk;
  }
}

// ---------------------------------------------------------------------------
// Prep kernel: build dense normalized adjacency M[d][s] (bf16, [160][160]) once.
//   deg = 1 (self loop) + in-degree; dinv = deg^-1/2; M[col][row] += dinv*dinv
// ---------------------------------------------------------------------------
extern __shared__ char smem_raw[];

__global__ __launch_bounds__(256) void build_adj(const int* __restrict__ ei,
                                                 unsigned short* __restrict__ Mg)
{
  __shared__ float deg[NN];
  __shared__ float dinv[NN];
  float* Ms = (float*)smem_raw;                    // [160*160] fp32
  const int tid = threadIdx.x;

  for (int i = tid; i < MROWS * MROWS; i += 256) Ms[i] = 0.f;
  for (int i = tid; i < NN; i += 256) deg[i] = 1.f;          // self loop
  __syncthreads();
  for (int e = tid; e < EE; e += 256) atomicAdd(&deg[ei[EE + e]], 1.f);
  __syncthreads();
  for (int i = tid; i < NN; i += 256) dinv[i] = rsqrtf(deg[i]);
  __syncthreads();
  for (int e = tid; e < EE; e += 256) {
    const int r = ei[e], c = ei[EE + e];
    atomicAdd(&Ms[c * MROWS + r], dinv[r] * dinv[c]);
  }
  for (int n = tid; n < NN; n += 256)
    atomicAdd(&Ms[n * MROWS + n], dinv[n] * dinv[n]);        // self loops
  __syncthreads();
  for (int i = tid; i < MROWS * MROWS; i += 256) Mg[i] = f2bf(Ms[i]);
}

// ---------------------------------------------------------------------------
// Fused 3-layer GCN: one workgroup per batch element, 256 threads (8 waves).
// ---------------------------------------------------------------------------
__global__ __launch_bounds__(256, 1) void gcn_fused(
    const float* __restrict__ x, const unsigned short* __restrict__ Mg,
    const float* __restrict__ W0, const float* __restrict__ W1, const float* __restrict__ W2,
    const float* __restrict__ b0, const float* __restrict__ b1, const float* __restrict__ b2,
    const float* __restrict__ bnw0, const float* __restrict__ bnb0,
    const float* __restrict__ bnw1, const float* __restrict__ bnb1,
    float* __restrict__ out)
{
  char* smem = smem_raw;
  const int b   = blockIdx.x;
  const int tid = threadIdx.x;

  // Adjacency bf16 (persistent) -- 51200B via 16B copies (L2-resident source)
  {
    const uint4* src = (const uint4*)Mg;
    uint4* dst = (uint4*)(smem + OFF_M);
    for (int i = tid; i < (MROWS * MROWS * 2) / 16; i += 256) dst[i] = src[i];
  }
  // x[b] slice -> bf16 A0 [160][320] (zero-padded rows/cols), 4-wide chunks.
  // This is the kernel's entire HBM read stream: keep it lean (one index calc
  // + 4 coalesced dword loads + 1 ds_store_b64 per 4 elements).
  {
    unsigned short* A = (unsigned short*)(smem + OFF_A);
    const float* xb = x + (size_t)b * NN * CIN;
    for (int i = tid; i < MROWS * (K0P / 4); i += 256) {
      const int n  = i / (K0P / 4);
      const int k0 = (i - n * (K0P / 4)) << 2;
      float v0 = 0.f, v1 = 0.f, v2 = 0.f, v3 = 0.f;
      if (n < NN) {
        const float* src = xb + n * CIN + k0;
        v0 = (k0 + 0 < CIN) ? src[0] : 0.f;
        v1 = (k0 + 1 < CIN) ? src[1] : 0.f;
        v2 = (k0 + 2 < CIN) ? src[2] : 0.f;
        v3 = (k0 + 3 < CIN) ? src[3] : 0.f;
      }
      uint2 pk;
      pk.x = pack2bf(v0, v1);
      pk.y = pack2bf(v2, v3);
      *(uint2*)(A + n * K0P + k0) = pk;
    }
  }
  // W0^T -> bf16 [128][320], 4-wide chunks (L2-resident source)
  {
    unsigned short* Wt = (unsigned short*)(smem + OFF_W);
    for (int i = tid; i < CH * (K0P / 4); i += 256) {
      const int n  = i / (K0P / 4);
      const int k0 = (i - n * (K0P / 4)) << 2;
      uint2 pk;
      pk.x = pack2bf((k0 + 0 < CIN) ? W0[(k0 + 0) * CH + n] : 0.f,
                     (k0 + 1 < CIN) ? W0[(k0 + 1) * CH + n] : 0.f);
      pk.y = pack2bf((k0 + 2 < CIN) ? W0[(k0 + 2) * CH + n] : 0.f,
                     (k0 + 3 < CIN) ? W0[(k0 + 3) * CH + n] : 0.f);
      *(uint2*)(Wt + n * K0P + k0) = pk;
    }
  }
  __syncthreads();

  // ---- Layer 0 ----
  wmma_gemm<0>(smem, OFF_A, K0P * 2, OFF_W, K0P * 2, K0P, nullptr, nullptr, nullptr, nullptr);
  __syncthreads();
  wmma_gemm<1>(smem, OFF_M, KAGG * 2, OFF_H, HT_LD * 2, KAGG, bnw0, bnb0, b0, nullptr);
  __syncthreads();

  // ---- Layer 1 ----
  stage_wt128(smem, W1);
  __syncthreads();
  wmma_gemm<0>(smem, OFF_A, CH * 2, OFF_W, CH * 2, CH, nullptr, nullptr, nullptr, nullptr);
  __syncthreads();
  wmma_gemm<1>(smem, OFF_M, KAGG * 2, OFF_H, HT_LD * 2, KAGG, bnw1, bnb1, b1, nullptr);
  __syncthreads();

  // ---- Layer 2 ----
  stage_wt128(smem, W2);
  __syncthreads();
  wmma_gemm<0>(smem, OFF_A, CH * 2, OFF_W, CH * 2, CH, nullptr, nullptr, nullptr, nullptr);
  __syncthreads();
  wmma_gemm<2>(smem, OFF_M, KAGG * 2, OFF_H, HT_LD * 2, KAGG, nullptr, nullptr, b2,
               out + (size_t)b * NN * CH);
}

// ---------------------------------------------------------------------------
extern "C" void kernel_launch(void* const* d_in, const int* in_sizes, int n_in,
                              void* d_out, int out_size, void* d_ws, size_t ws_size,
                              hipStream_t stream)
{
  (void)in_sizes; (void)n_in; (void)out_size; (void)ws_size;
  const float* x    = (const float*)d_in[0];
  const int*   ei   = (const int*)d_in[1];     // edge_index [2,E]
  const float* W0   = (const float*)d_in[2];
  const float* b0   = (const float*)d_in[3];
  const float* bnw0 = (const float*)d_in[4];
  const float* bnb0 = (const float*)d_in[5];
  const float* W1   = (const float*)d_in[6];
  const float* b1   = (const float*)d_in[7];
  const float* bnw1 = (const float*)d_in[8];
  const float* bnb1 = (const float*)d_in[9];
  const float* W2   = (const float*)d_in[10];
  const float* b2   = (const float*)d_in[11];
  float* out = (float*)d_out;

  unsigned short* Mg = (unsigned short*)d_ws;  // dense adjacency bf16 [160*160]

  build_adj<<<1, 256, MROWS * MROWS * sizeof(float), stream>>>(ei, Mg);
  gcn_fused<<<1024, 256, LDS_BYTES, stream>>>(x, Mg, W0, W1, W2, b0, b1, b2,
                                              bnw0, bnb0, bnw1, bnb1, out);
}
